// WoBObservationEncoder_9096740733414
// MI455X (gfx1250) — compile-verified, matrix operand order checked by reading, not run on hardware
//
#include <hip/hip_runtime.h>

// ---------------------------------------------------------------------------
// GraphSAGE 2-layer forward for MI455X (gfx1250), wave32 + WMMA f16.
//   N=50000 nodes, E=800000 edges, G=64 graphs
//   layer1: relu( mean_agg(h) @ W1l + b1 + h @ W1r ),  h = [x | votes]  (K=108)
//   layer2:       mean_agg(h1) @ W2l + b2 + h1 @ W2r                    (K=128)
//   out   : segment_max over batch -> [64, 64]
// Fusion: A = [agg | h] (K padded to 32), B = [Wl ; Wr]^T stored K-contiguous
// so both WMMA fragments are two aligned 16B loads per lane. Each wave now
// computes 32x64 of C (2 M-tiles x 4 N-tiles) so every B fragment feeds 2
// WMMAs: 12 b128 loads -> 8 v_wmma per 32-wide K step.
// ---------------------------------------------------------------------------

#define N_NODES 50000
#define N_EDGES 800000
#define N_GRAPHS 64
#define IN_DIM 108   // 107 x-features + 1 vote
#define HID 128
#define OUT_DIM 64
#define K1 224       // 108 + 108 padded to multiple of 32
#define K2 256       // 128 + 128

typedef __attribute__((ext_vector_type(16))) _Float16 v16h;
typedef __attribute__((ext_vector_type(8)))  _Float16 v8h;
typedef __attribute__((ext_vector_type(8)))  float    v8f;

// ---- generic fill ----------------------------------------------------------
__global__ void fill_f32(float* __restrict__ p, long long n, float v) {
  long long i = (long long)blockIdx.x * blockDim.x + threadIdx.x;
  if (i < n) p[i] = v;
}

// ---- weight transpose + f16 convert ---------------------------------------
// WT1[n*K1 + k] = k<108 ? W1l[k,n] : k<216 ? W1r[k-108,n] : 0   (n<128)
__global__ void wconv1_kernel(const float* __restrict__ W1l,
                              const float* __restrict__ W1r,
                              _Float16* __restrict__ WT1) {
  int i = blockIdx.x * blockDim.x + threadIdx.x;   // over 128*224
  if (i >= HID * K1) return;
  int n = i / K1, k = i % K1;
  float v = 0.f;
  if (k < IN_DIM)            v = W1l[k * HID + n];
  else if (k < 2 * IN_DIM)   v = W1r[(k - IN_DIM) * HID + n];
  WT1[n * K1 + k] = (_Float16)v;
}

// WT2[n*K2 + k] = k<128 ? W2l[k,n] : W2r[k-128,n]   (n<64)
__global__ void wconv2_kernel(const float* __restrict__ W2l,
                              const float* __restrict__ W2r,
                              _Float16* __restrict__ WT2) {
  int i = blockIdx.x * blockDim.x + threadIdx.x;   // over 64*256
  if (i >= OUT_DIM * K2) return;
  int n = i / K2, k = i % K2;
  float v = (k < HID) ? W2l[k * OUT_DIM + n] : W2r[(k - HID) * OUT_DIM + n];
  WT2[n * K2 + k] = (_Float16)v;
}

// ---- degree ----------------------------------------------------------------
__global__ void deg_kernel(const int* __restrict__ ei, float* __restrict__ deg) {
  int e = blockIdx.x * blockDim.x + threadIdx.x;
  if (e < N_EDGES) atomicAdd(&deg[ei[N_EDGES + e]], 1.0f);
}

// ---- scatter layer 1: agg1[dst] += h[src], h built on the fly from x|votes -
__global__ void scatter1_kernel(const int* __restrict__ ei,
                                const float* __restrict__ x,
                                const float* __restrict__ votes,
                                float* __restrict__ agg1) {
  int e = blockIdx.x;          // one edge per 128-thread block
  int d = threadIdx.x;
  if (d >= IN_DIM) return;
  int s = ei[e], t = ei[N_EDGES + e];
  float v = (d < IN_DIM - 1) ? x[(long long)s * (IN_DIM - 1) + d] : votes[s];
  atomicAdd(&agg1[(long long)t * IN_DIM + d], v);
}

// ---- build A1 = [agg1/deg | x|votes | 0pad] as f16 [N, 224] ----------------
__global__ void hcat1_kernel(const float* __restrict__ agg1,
                             const float* __restrict__ deg,
                             const float* __restrict__ x,
                             const float* __restrict__ votes,
                             _Float16* __restrict__ hcat1) {
  int n = blockIdx.x;
  int k = threadIdx.x;         // 224 threads
  float v = 0.f;
  if (k < IN_DIM) {
    float dg = deg[n];
    v = agg1[(long long)n * IN_DIM + k] / fmaxf(dg, 1.0f);
  } else if (k < 2 * IN_DIM) {
    int j = k - IN_DIM;
    v = (j < IN_DIM - 1) ? x[(long long)n * (IN_DIM - 1) + j] : votes[n];
  }
  hcat1[(long long)n * K1 + k] = (_Float16)v;
}

// ---- scatter layer 2: agg2[dst] += h1[src] ---------------------------------
__global__ void scatter2_kernel(const int* __restrict__ ei,
                                const float* __restrict__ h1,
                                float* __restrict__ agg2) {
  int e = blockIdx.x;
  int d = threadIdx.x;         // 128
  int s = ei[e], t = ei[N_EDGES + e];
  atomicAdd(&agg2[(long long)t * HID + d], h1[(long long)s * HID + d]);
}

// ---- normalize agg2 into first half of A2 ----------------------------------
__global__ void hcat2_kernel(const float* __restrict__ agg2,
                             const float* __restrict__ deg,
                             _Float16* __restrict__ hcat2) {
  int n = blockIdx.x;
  int k = threadIdx.x;         // 128
  float dg = deg[n];
  float v = agg2[(long long)n * HID + k] / fmaxf(dg, 1.0f);
  hcat2[(long long)n * K2 + k] = (_Float16)v;
}

// ---- WMMA fragment load: two aligned 16B chunks per lane -------------------
// 16-bit A layout (ISA 7.12.2): lane l (0..15) row M=l holds K = hi*8+{0..7}
// and 16+hi*8+{0..7} where hi = lane>=16. B^T stored K-contiguous uses the
// identical pattern with row = output column.
__device__ inline v16h load_frag(const _Float16* __restrict__ base, int ld,
                                 int row, int k0, int hi) {
  const _Float16* p = base + (long long)row * ld + k0 + hi * 8;
  v8h lo = *(const v8h*)(p);
  v8h hi8 = *(const v8h*)(p + 16);
  v16h r;
#pragma unroll
  for (int i = 0; i < 8; ++i) { r[i] = lo[i]; r[i + 8] = hi8[i]; }
  return r;
}

#define WMMA_F16(A_, B_, C_) \
  __builtin_amdgcn_wmma_f32_16x16x32_f16(false, (A_), false, (B_), (short)0, (C_), false, false)

// ---- GEMM1: [N,224]f16 @ [224,128]f16 -> relu(+b1) -> h1 f32, hcat2 hi-half
// block = 128 threads (4 waves); wave computes 32 rows x 64 cols
__global__ void gemm1_kernel(const _Float16* __restrict__ A,
                             const _Float16* __restrict__ BT,
                             const float* __restrict__ bias,
                             float* __restrict__ h1,
                             _Float16* __restrict__ hcat2) {
  int wave = threadIdx.x >> 5;
  int lane = threadIdx.x & 31;
  int l = lane & 15, hi = lane >> 4;
  int tileM = blockIdx.x * 128 + wave * 32;       // 2 M-tiles per wave
  int tileN = blockIdx.y * 64;
  int arow0 = tileM + l;      if (arow0 >= N_NODES) arow0 = N_NODES - 1;
  int arow1 = tileM + 16 + l; if (arow1 >= N_NODES) arow1 = N_NODES - 1;

  v8f acc[2][4];
#pragma unroll
  for (int m = 0; m < 2; ++m)
#pragma unroll
    for (int t = 0; t < 4; ++t) acc[m][t] = (v8f)0.0f;

#pragma unroll
  for (int k0 = 0; k0 < K1; k0 += 32) {
    v16h a0 = load_frag(A, K1, arow0, k0, hi);
    v16h a1 = load_frag(A, K1, arow1, k0, hi);
#pragma unroll
    for (int t = 0; t < 4; ++t) {
      v16h b = load_frag(BT, K1, tileN + t * 16 + l, k0, hi);
      acc[0][t] = WMMA_F16(a0, b, acc[0][t]);
      acc[1][t] = WMMA_F16(a1, b, acc[1][t]);
    }
  }

#pragma unroll
  for (int m = 0; m < 2; ++m)
#pragma unroll
    for (int t = 0; t < 4; ++t) {
      int col = tileN + t * 16 + l;
      float b = bias[col];
#pragma unroll
      for (int r = 0; r < 8; ++r) {
        int row = tileM + m * 16 + r + hi * 8;
        if (row < N_NODES) {
          float v = fmaxf(acc[m][t][r] + b, 0.0f);
          h1[(long long)row * HID + col] = v;
          hcat2[(long long)row * K2 + HID + col] = (_Float16)v;  // fused staging
        }
      }
    }
}

// ---- float atomic max via read-check + CAS ---------------------------------
__device__ inline void atomicMaxF(float* addr, float val) {
  int* ai = (int*)addr;
  int old = __float_as_int(*addr);
  while (__int_as_float(old) < val) {
    int assumed = old;
    old = atomicCAS(ai, assumed, __float_as_int(val));
    if (old == assumed) break;
  }
}

// ---- GEMM2: [N,256]f16 @ [256,64]f16 + b2, fused segment_max -> d_out ------
__global__ void gemm2_kernel(const _Float16* __restrict__ A,
                             const _Float16* __restrict__ BT,
                             const float* __restrict__ bias,
                             const int* __restrict__ batch,
                             float* __restrict__ out) {
  int wave = threadIdx.x >> 5;
  int lane = threadIdx.x & 31;
  int l = lane & 15, hi = lane >> 4;
  int tileM = blockIdx.x * 128 + wave * 32;
  int arow0 = tileM + l;      if (arow0 >= N_NODES) arow0 = N_NODES - 1;
  int arow1 = tileM + 16 + l; if (arow1 >= N_NODES) arow1 = N_NODES - 1;

  v8f acc[2][4];
#pragma unroll
  for (int m = 0; m < 2; ++m)
#pragma unroll
    for (int t = 0; t < 4; ++t) acc[m][t] = (v8f)0.0f;

#pragma unroll
  for (int k0 = 0; k0 < K2; k0 += 32) {
    v16h a0 = load_frag(A, K2, arow0, k0, hi);
    v16h a1 = load_frag(A, K2, arow1, k0, hi);
#pragma unroll
    for (int t = 0; t < 4; ++t) {
      v16h b = load_frag(BT, K2, t * 16 + l, k0, hi);
      acc[0][t] = WMMA_F16(a0, b, acc[0][t]);
      acc[1][t] = WMMA_F16(a1, b, acc[1][t]);
    }
  }

#pragma unroll
  for (int m = 0; m < 2; ++m)
#pragma unroll
    for (int t = 0; t < 4; ++t) {
      int col = t * 16 + l;
      float b = bias[col];
#pragma unroll
      for (int r = 0; r < 8; ++r) {
        int row = tileM + m * 16 + r + hi * 8;
        if (row < N_NODES) {
          float v = acc[m][t][r] + b;
          int g = batch[row];
          atomicMaxF(&out[g * OUT_DIM + col], v);
        }
      }
    }
}

// ---------------------------------------------------------------------------
extern "C" void kernel_launch(void* const* d_in, const int* in_sizes, int n_in,
                              void* d_out, int out_size, void* d_ws, size_t ws_size,
                              hipStream_t stream) {
  const float* x     = (const float*)d_in[0];   // [N, 107]
  const float* votes = (const float*)d_in[1];   // [N, 1]
  const int*   ei    = (const int*)d_in[2];     // [2, E]
  const int*   batch = (const int*)d_in[3];     // [N]
  const float* W1l   = (const float*)d_in[4];   // [108,128]
  const float* b1    = (const float*)d_in[5];   // [128]
  const float* W1r   = (const float*)d_in[6];   // [108,128]
  const float* W2l   = (const float*)d_in[7];   // [128,64]
  const float* b2    = (const float*)d_in[8];   // [64]
  const float* W2r   = (const float*)d_in[9];   // [128,64]
  float* out = (float*)d_out;                   // [64,64]

  // workspace carving (256B aligned)
  char* ws = (char*)d_ws;
  size_t off = 0;
  auto carve = [&](size_t bytes) { void* p = ws + off;
    off = (off + bytes + 255) & ~(size_t)255; return p; };
  float*    agg1  = (float*)   carve((size_t)N_NODES * IN_DIM * 4);
  float*    agg2  = (float*)   carve((size_t)N_NODES * HID * 4);
  float*    deg   = (float*)   carve((size_t)N_NODES * 4);
  float*    h1    = (float*)   carve((size_t)N_NODES * HID * 4);
  _Float16* hcat1 = (_Float16*)carve((size_t)N_NODES * K1 * 2);
  _Float16* hcat2 = (_Float16*)carve((size_t)N_NODES * K2 * 2);
  _Float16* WT1   = (_Float16*)carve((size_t)HID * K1 * 2);
  _Float16* WT2   = (_Float16*)carve((size_t)OUT_DIM * K2 * 2);
  (void)ws_size; (void)in_sizes; (void)n_in; (void)out_size;

  // 1) zero scratch, init output to -FLT_MAX for max-reduce
  {
    long long n1 = (long long)N_NODES * IN_DIM;
    long long n2 = (long long)N_NODES * HID;
    fill_f32<<<dim3((unsigned)((n1 + 255) / 256)), 256, 0, stream>>>(agg1, n1, 0.f);
    fill_f32<<<dim3((unsigned)((n2 + 255) / 256)), 256, 0, stream>>>(agg2, n2, 0.f);
    fill_f32<<<dim3((N_NODES + 255) / 256), 256, 0, stream>>>(deg, N_NODES, 0.f);
    fill_f32<<<dim3((N_GRAPHS * OUT_DIM + 255) / 256), 256, 0, stream>>>(
        out, N_GRAPHS * OUT_DIM, -3.402823466e38f);
  }

  // 2) transpose + f16-convert the fused weight matrices
  wconv1_kernel<<<dim3((HID * K1 + 255) / 256), 256, 0, stream>>>(W1l, W1r, WT1);
  wconv2_kernel<<<dim3((OUT_DIM * K2 + 255) / 256), 256, 0, stream>>>(W2l, W2r, WT2);

  // 3) degrees + layer-1 scatter-sum
  deg_kernel<<<dim3((N_EDGES + 255) / 256), 256, 0, stream>>>(ei, deg);
  scatter1_kernel<<<dim3(N_EDGES), 128, 0, stream>>>(ei, x, votes, agg1);

  // 4) stage A1 = [agg1/deg | x|votes | 0] in f16
  hcat1_kernel<<<dim3(N_NODES), K1, 0, stream>>>(agg1, deg, x, votes, hcat1);

  // 5) GEMM1 (WMMA), fused bias+relu, writes h1 f32 + hcat2 upper half f16
  gemm1_kernel<<<dim3((N_NODES + 127) / 128, HID / 64), 128, 0, stream>>>(
      hcat1, WT1, b1, h1, hcat2);

  // 6) layer-2 scatter-sum + normalize into hcat2 lower half
  scatter2_kernel<<<dim3(N_EDGES), HID, 0, stream>>>(ei, h1, agg2);
  hcat2_kernel<<<dim3(N_NODES), HID, 0, stream>>>(agg2, deg, hcat2);

  // 7) GEMM2 (WMMA) with fused per-graph atomic max into d_out
  gemm2_kernel<<<dim3((N_NODES + 127) / 128), 128, 0, stream>>>(
      hcat2, WT2, b2, batch, out);
}
